// MultiHeadFeatureInteraction_41592463294927
// MI455X (gfx1250) — compile-verified, matrix operand order checked by reading, not run on hardware
//
#include <hip/hip_runtime.h>
#include <hip/hip_bf16.h>

// ---------------------------------------------------------------------------
// MultiHeadFeatureInteraction on MI455X (gfx1250), wave32 + WMMA f16.
//   B=2, S=2048, DIM=1024, H=16, Hd=64, SCALE = 1/8
// Pipeline: f32->f16 casts -> 3x WMMA GEMM (Q,K,V; V transposed) ->
//           flash-attention (WMMA QK^T + PV, online softmax w/ DPP reductions)
//           -> WMMA GEMM output projection (f32 out).
// ---------------------------------------------------------------------------

typedef __attribute__((ext_vector_type(16))) _Float16 v16h;
typedef __attribute__((ext_vector_type(8)))  _Float16 v8h;
typedef __attribute__((ext_vector_type(8)))  float    v8f;

#define MHA_B    2
#define MHA_S    2048
#define MHA_D    1024
#define MHA_H    16
#define MHA_HD   64
#define MHA_BS   (MHA_B * MHA_S)      // 4096 rows
#define MHA_SCALE 0.125f              // 1/sqrt(64)

// ---------------------------------------------------------------------------
// Fragment loader for 16-bit A/B matrices (16x32 MxK / 32x16 KxN as columns).
// ISA layout (CDNA5 ISA, 16-bit A-matrix 16x32):
//   lane L: row = L&15, K-half kh = (L>>4)*8
//   element e: K = kh + e        (e in 0..7)
//              K = kh + 8 + e    (e in 8..15)  i.e. kh+16 .. kh+23
// Both X (A) and W (B columns) are row-major K-contiguous => same loader.
// ---------------------------------------------------------------------------
__device__ __forceinline__ v16h load_frag16(const _Float16* __restrict__ base, int ld) {
    const int lane = threadIdx.x & 31;
    const int row  = lane & 15;
    const int kh   = (lane >> 4) << 3;           // 0 or 8
    const _Float16* p = base + row * ld + kh;
    v8h lo = *(const v8h*)(p);                   // K = kh .. kh+7
    v8h hi = *(const v8h*)(p + 16);              // K = kh+16 .. kh+23
    v16h f;
#pragma unroll
    for (int i = 0; i < 8; ++i) { f[i] = lo[i]; f[i + 8] = hi[i]; }
    return f;
}

__device__ __forceinline__ v8f wmma_f16(v16h a, v16h b, v8f c) {
    return __builtin_amdgcn_wmma_f32_16x16x32_f16(
        /*neg_a=*/false, a, /*neg_b=*/false, b,
        /*c_mod=*/(short)0, c, /*reuse_a=*/false, /*reuse_b=*/false);
}

// ---------------------------------------------------------------------------
// DPP cross-lane: pure-VALU reductions within each 16-lane row (DPP "row"),
// matching the WMMA C-layout where lanes 0-15 / 16-31 hold disjoint M rows.
//   0xB1  = quad_perm [1,0,3,2]  (xor 1)
//   0x4E  = quad_perm [2,3,0,1]  (xor 2)
//   0x141 = row_half_mirror      (xor 4 after quads reduced)
//   0x140 = row_mirror           (xor 8 after halves reduced)
// ---------------------------------------------------------------------------
template <int CTRL>
__device__ __forceinline__ float dpp_bcast(float x) {
    return __int_as_float(__builtin_amdgcn_update_dpp(
        0, __float_as_int(x), CTRL, 0xF, 0xF, true));
}

__device__ __forceinline__ float row_max16(float x) {
    x = fmaxf(x, dpp_bcast<0xB1>(x));
    x = fmaxf(x, dpp_bcast<0x4E>(x));
    x = fmaxf(x, dpp_bcast<0x141>(x));
    x = fmaxf(x, dpp_bcast<0x140>(x));
    return x;
}

__device__ __forceinline__ float row_sum16(float x) {
    x += dpp_bcast<0xB1>(x);
    x += dpp_bcast<0x4E>(x);
    x += dpp_bcast<0x141>(x);
    x += dpp_bcast<0x140>(x);
    return x;
}

// ---------------------------------------------------------------------------
// f32 -> f16 elementwise cast
// ---------------------------------------------------------------------------
__global__ void cvt_f32_f16_kernel(const float* __restrict__ in,
                                   _Float16* __restrict__ out, int n) {
    int i = blockIdx.x * blockDim.x + threadIdx.x;
    if (i < n) out[i] = (_Float16)in[i];
}

// ---------------------------------------------------------------------------
// GEMM: out = A[M,K] * W[N,K]^T + bias[N]      (nn.Linear NT layout)
// Block: 256 threads = 8 waves, block tile 64(M) x 256(N),
// wave tile 32x64 = 2x4 WMMA 16x16 tiles, K-step 32.
//   Per K-step: 6 fragment loads -> 8 WMMAs (1.5 b128 loads per WMMA).
// OUT_MODE: 0 = f16 row-major [M,N]   (Q, K projections)
//           1 = f16 transposed [N,M]  (V projection -> V^T)
//           2 = f32 row-major [M,N]   (output projection -> d_out)
// ---------------------------------------------------------------------------
template <int OUT_MODE>
__global__ void gemm_bias_wmma_kernel(const _Float16* __restrict__ A,
                                      const _Float16* __restrict__ W,
                                      const float* __restrict__ bias,
                                      void* __restrict__ out,
                                      int M, int N, int K) {
    const int wid  = threadIdx.x >> 5;           // 0..7
    const int lane = threadIdx.x & 31;
    const int wrow = wid & 1;                    // 2 wave-rows
    const int wcol = wid >> 1;                   // 4 wave-cols
    const int row0 = blockIdx.y * 64 + wrow * 32;
    const int col0 = blockIdx.x * 256 + wcol * 64;

    v8f acc[2][4];
#pragma unroll
    for (int i = 0; i < 2; ++i)
#pragma unroll
        for (int j = 0; j < 4; ++j) acc[i][j] = (v8f){};

    const _Float16* arow0 = A + (size_t)row0 * K;
    const _Float16* arow1 = A + (size_t)(row0 + 16) * K;

    for (int kk = 0; kk < K; kk += 32) {
        // Pull the next K-tile toward the caches (global_prefetch_b8).
        if (kk + 32 < K) {
            __builtin_prefetch(arow0 + kk + 32 + (size_t)(lane & 15) * K, 0, 1);
            __builtin_prefetch(W + (size_t)(col0 + (lane & 15)) * K + kk + 32, 0, 1);
        }
        v16h a0 = load_frag16(arow0 + kk, K);
        v16h a1 = load_frag16(arow1 + kk, K);
#pragma unroll
        for (int j = 0; j < 4; ++j) {
            v16h b = load_frag16(W + (size_t)(col0 + j * 16) * K + kk, K);
            acc[0][j] = wmma_f16(a0, b, acc[0][j]);
            acc[1][j] = wmma_f16(a1, b, acc[1][j]);
        }
    }

    // C/D layout: lane holds column n = lane&15; VGPR v -> row m = v + 8*(lane>>4)
    const int n  = lane & 15;
    const int mh = lane >> 4;
#pragma unroll
    for (int ti = 0; ti < 2; ++ti) {
#pragma unroll
        for (int tj = 0; tj < 4; ++tj) {
            const int c = col0 + tj * 16 + n;
            const float bval = bias[c];
#pragma unroll
            for (int v = 0; v < 8; ++v) {
                const int r = row0 + ti * 16 + v + 8 * mh;
                const float val = acc[ti][tj][v] + bval;
                if (OUT_MODE == 0) {
                    ((_Float16*)out)[(size_t)r * N + c] = (_Float16)val;
                } else if (OUT_MODE == 1) {
                    ((_Float16*)out)[(size_t)c * M + r] = (_Float16)val;
                } else {
                    ((float*)out)[(size_t)r * N + c] = val;
                }
            }
        }
    }
}

// ---------------------------------------------------------------------------
// Flash attention.
// Q, K: f16 [BS, D] row-major.  Vt: f16 [D, BS] (transposed).  ctx: f16 [BS, D].
// Grid: (S/128, B*H). Block: 256 threads = 8 independent waves; each wave owns
// 16 q-rows and a private LDS slab (no workgroup barriers needed: per-wave LDS
// ops are executed in order; __builtin_amdgcn_wave_barrier() pins compiler
// ordering around the C-layout -> A-layout P transpose).
// Row-wise softmax stats reduced with DPP (no ds_bpermute round-trips).
// ---------------------------------------------------------------------------
__global__ void flash_attn_kernel(const _Float16* __restrict__ Q,
                                  const _Float16* __restrict__ Km,
                                  const _Float16* __restrict__ Vt,
                                  _Float16* __restrict__ ctx) {
    __shared__ __align__(16) _Float16 ldsP[8][16 * 32];

    const int w    = threadIdx.x >> 5;
    const int lane = threadIdx.x & 31;
    const int bh   = blockIdx.y;
    const int b    = bh >> 4;
    const int h    = bh & 15;
    const int hoff = h * MHA_HD;

    const int qrow = b * MHA_S + blockIdx.x * 128 + w * 16;  // global row of q tile

    const v16h qa0 = load_frag16(Q + (size_t)qrow * MHA_D + hoff,      MHA_D);
    const v16h qa1 = load_frag16(Q + (size_t)qrow * MHA_D + hoff + 32, MHA_D);

    float m_i[8], l_i[8];
    v8f o[4];
#pragma unroll
    for (int v = 0; v < 8; ++v) { m_i[v] = -1.0e30f; l_i[v] = 0.0f; }
#pragma unroll
    for (int t = 0; t < 4; ++t) o[t] = (v8f){};

    const int n  = lane & 15;
    const int mh = lane >> 4;

    for (int kb = 0; kb < MHA_S; kb += 32) {
        const int krow = b * MHA_S + kb;

        // scores: keys kb..kb+15 (c0) and kb+16..kb+31 (c1)
        v16h k00 = load_frag16(Km + (size_t)krow * MHA_D + hoff,             MHA_D);
        v16h k01 = load_frag16(Km + (size_t)krow * MHA_D + hoff + 32,        MHA_D);
        v16h k10 = load_frag16(Km + (size_t)(krow + 16) * MHA_D + hoff,      MHA_D);
        v16h k11 = load_frag16(Km + (size_t)(krow + 16) * MHA_D + hoff + 32, MHA_D);
        v8f c0 = (v8f){};
        c0 = wmma_f16(qa0, k00, c0);
        c0 = wmma_f16(qa1, k01, c0);
        v8f c1 = (v8f){};
        c1 = wmma_f16(qa0, k10, c1);
        c1 = wmma_f16(qa1, k11, c1);

        // online softmax; row m = v + 8*mh lives across the 16-lane half
        float alpha[8], p0[8], p1[8];
#pragma unroll
        for (int v = 0; v < 8; ++v) {
            float s0 = c0[v] * MHA_SCALE;
            float s1 = c1[v] * MHA_SCALE;
            float t = row_max16(fmaxf(s0, s1));
            const float mnew = fmaxf(m_i[v], t);
            alpha[v] = __expf(m_i[v] - mnew);
            p0[v] = __expf(s0 - mnew);
            p1[v] = __expf(s1 - mnew);
            const float r = row_sum16(p0[v] + p1[v]);
            l_i[v] = l_i[v] * alpha[v] + r;
            m_i[v] = mnew;
        }
#pragma unroll
        for (int t = 0; t < 4; ++t)
#pragma unroll
            for (int v = 0; v < 8; ++v) o[t][v] *= alpha[v];

        // P (C layout) -> per-wave LDS slab -> A layout.
        // Same-wave DS ops execute in order; only compiler ordering is needed.
        __builtin_amdgcn_wave_barrier();
#pragma unroll
        for (int v = 0; v < 8; ++v) {
            ldsP[w][(v + 8 * mh) * 32 + n]      = (_Float16)p0[v];
            ldsP[w][(v + 8 * mh) * 32 + 16 + n] = (_Float16)p1[v];
        }
        __builtin_amdgcn_wave_barrier();
        v16h pfrag;
        {
            const int row = lane & 15;
            const int kh  = (lane >> 4) << 3;
            const _Float16* pp = &ldsP[w][row * 32 + kh];
            v8h lo = *(const v8h*)(pp);
            v8h hi = *(const v8h*)(pp + 16);
#pragma unroll
            for (int i = 0; i < 8; ++i) { pfrag[i] = lo[i]; pfrag[i + 8] = hi[i]; }
        }
        __builtin_amdgcn_wave_barrier();

        // O += P(16x32) x V(32x64); V^T rows are K-contiguous
#pragma unroll
        for (int t = 0; t < 4; ++t) {
            v16h vf = load_frag16(Vt + (size_t)(hoff + t * 16) * MHA_BS + krow, MHA_BS);
            o[t] = wmma_f16(pfrag, vf, o[t]);
        }
    }

    // normalize and write context [BS, D]
#pragma unroll
    for (int t = 0; t < 4; ++t) {
#pragma unroll
        for (int v = 0; v < 8; ++v) {
            const int r = qrow + v + 8 * mh;
            const int c = hoff + t * 16 + n;
            ctx[(size_t)r * MHA_D + c] = (_Float16)(o[t][v] / l_i[v]);
        }
    }
}

// ---------------------------------------------------------------------------
// Host launcher
// ---------------------------------------------------------------------------
extern "C" void kernel_launch(void* const* d_in, const int* in_sizes, int n_in,
                              void* d_out, int out_size, void* d_ws, size_t ws_size,
                              hipStream_t stream) {
    const float* x  = (const float*)d_in[0];
    const float* Wq = (const float*)d_in[1];
    const float* bq = (const float*)d_in[2];
    const float* Wk = (const float*)d_in[3];
    const float* bk = (const float*)d_in[4];
    const float* Wv = (const float*)d_in[5];
    const float* bv = (const float*)d_in[6];
    const float* Wo = (const float*)d_in[7];
    const float* bo = (const float*)d_in[8];

    const int M = MHA_BS;   // 4096
    const int N = MHA_D;    // 1024
    const int K = MHA_D;    // 1024
    const size_t XE = (size_t)M * K;   // 4 Mi elems
    const size_t WE = (size_t)N * K;   // 1 Mi elems

    _Float16* ws  = (_Float16*)d_ws;
    _Float16* xh  = ws;
    _Float16* wqh = xh  + XE;
    _Float16* wkh = wqh + WE;
    _Float16* wvh = wkh + WE;
    _Float16* woh = wvh + WE;
    _Float16* Qh  = woh + WE;
    _Float16* Kh  = Qh  + XE;
    _Float16* Vt  = Kh  + XE;   // transposed: [D, BS]
    _Float16* ctx = Vt  + XE;
    // total: 5*XE + 4*WE = 24 Mi halves = 48 MiB of workspace

    const int CT = 256;
    cvt_f32_f16_kernel<<<(int)((XE + CT - 1) / CT), CT, 0, stream>>>(x,  xh,  (int)XE);
    cvt_f32_f16_kernel<<<(int)((WE + CT - 1) / CT), CT, 0, stream>>>(Wq, wqh, (int)WE);
    cvt_f32_f16_kernel<<<(int)((WE + CT - 1) / CT), CT, 0, stream>>>(Wk, wkh, (int)WE);
    cvt_f32_f16_kernel<<<(int)((WE + CT - 1) / CT), CT, 0, stream>>>(Wv, wvh, (int)WE);
    cvt_f32_f16_kernel<<<(int)((WE + CT - 1) / CT), CT, 0, stream>>>(Wo, woh, (int)WE);

    dim3 ggrid(N / 256, M / 64);   // (4, 64)
    gemm_bias_wmma_kernel<0><<<ggrid, 256, 0, stream>>>(xh, wqh, bq, (void*)Qh, M, N, K);
    gemm_bias_wmma_kernel<0><<<ggrid, 256, 0, stream>>>(xh, wkh, bk, (void*)Kh, M, N, K);
    gemm_bias_wmma_kernel<1><<<ggrid, 256, 0, stream>>>(xh, wvh, bv, (void*)Vt, M, N, K);

    dim3 agrid(MHA_S / 128, MHA_B * MHA_H);   // (16, 32)
    flash_attn_kernel<<<agrid, 256, 0, stream>>>(Qh, Kh, Vt, ctx);

    gemm_bias_wmma_kernel<2><<<ggrid, 256, 0, stream>>>(ctx, woh, bo, d_out, M, N, K);
}
